// DCRNNDecoder_66133906423963
// MI455X (gfx1250) — compile-verified, matrix operand order checked by reading, not run on hardware
//
#include <hip/hip_runtime.h>
#include <hip/hip_bf16.h>
#include <math.h>

typedef __bf16 bf16;
typedef __attribute__((ext_vector_type(16))) __bf16 v16bf;
typedef __attribute__((ext_vector_type(8)))  __bf16 v8bf;
typedef __attribute__((ext_vector_type(8)))  float  v8f;
typedef __attribute__((ext_vector_type(2)))  float  v2f;
typedef __attribute__((ext_vector_type(4)))  int    i32x4;

#define AS1 __attribute__((address_space(1)))
#define AS3 __attribute__((address_space(3)))

#define NB 64     // batch
#define TT 12     // time steps
#define NN 1000   // nodes
#define NP 1024   // padded nodes
#define HH 64     // hidden
#define MM 4      // supports
#define LDA 1032  // padded LDS row stride (bf16 elems): 2064B -> lanes hit distinct banks

__device__ __forceinline__ float sigmoidf_(float x) { return 1.0f / (1.0f + __expf(-x)); }

// Load a 16-element bf16 WMMA fragment: two contiguous 16B halves (k..k+7, k+16..k+23)
__device__ __forceinline__ v16bf load_frag(const bf16* p) {
    v8bf lo = *(const v8bf*)p;
    v8bf hi = *(const v8bf*)(p + 16);
    v16bf r;
#pragma unroll
    for (int i = 0; i < 8; ++i) { r[i] = lo[i]; r[i + 8] = hi[i]; }
    return r;
}

// ---------------------------------------------------------------------------
// Convert supports [M,1000,1000] fp32 -> padded bf16 [M,1024,1024] (zero pad)
// ---------------------------------------------------------------------------
__global__ void pack_support(const float* __restrict__ sup, bf16* __restrict__ Abf) {
    size_t idx = (size_t)blockIdx.x * 256 + threadIdx.x;   // over 4*1024*1024
    int m = (int)(idx >> 20);
    int rem = (int)(idx & ((1u << 20) - 1));
    int i = rem >> 10, j = rem & 1023;
    float v = 0.0f;
    if (i < NN && j < NN) v = sup[(size_t)m * NN * NN + (size_t)i * NN + j];
    Abf[idx] = (bf16)v;
}

// ---------------------------------------------------------------------------
// init_hidden [2,B,N,H] -> h0/h1 stored as [N][B][H]
// ---------------------------------------------------------------------------
__global__ void init_h(const float* __restrict__ ih, float* __restrict__ h0, float* __restrict__ h1) {
    size_t idx = (size_t)blockIdx.x * 256 + threadIdx.x;   // over 2*64*1000*64
    int h = (int)(idx & 63);
    int n = (int)((idx >> 6) % NN);
    int b = (int)((idx / ((size_t)NN * 64)) % NB);
    int l = (int)(idx / ((size_t)NB * NN * 64));
    float* dst = (l == 0) ? h0 : h1;
    dst[((size_t)n * NB + b) * HH + h] = ih[idx];
}

// ---------------------------------------------------------------------------
// Pack Xt[col][k] bf16, col = b*65+f (layer0): f==0 -> input[b,t,n], else h0
// ---------------------------------------------------------------------------
__global__ void pack_x0(const float* __restrict__ input, const float* __restrict__ h0,
                        bf16* __restrict__ Xt, int t) {
    size_t idx = (size_t)blockIdx.x * 256 + threadIdx.x;   // over 4160*1024
    int n = (int)(idx & 1023);
    int col = (int)(idx >> 10);
    if (col >= 65 * NB) return;
    int b = col / 65, f = col - b * 65;
    float v = 0.0f;
    if (n < NN)
        v = (f == 0) ? input[((size_t)b * TT + t) * NN + n]
                     : h0[((size_t)n * NB + b) * HH + (f - 1)];
    Xt[(size_t)col * NP + n] = (bf16)v;
}

// Pack Xt for layer1: col = b*128+f, f<64 -> h0, else h1
__global__ void pack_x1(const float* __restrict__ h0, const float* __restrict__ h1,
                        bf16* __restrict__ Xt) {
    size_t idx = (size_t)blockIdx.x * 256 + threadIdx.x;   // over 8192*1024
    int n = (int)(idx & 1023);
    int col = (int)(idx >> 10);
    int b = col >> 7, f = col & 127;
    float v = 0.0f;
    if (n < NN)
        v = (f < HH) ? h0[((size_t)n * NB + b) * HH + f]
                     : h1[((size_t)n * NB + b) * HH + (f - HH)];
    Xt[(size_t)col * NP + n] = (bf16)v;
}

// ---------------------------------------------------------------------------
// Diffusion GEMM (bf16 WMMA). Block = 8 waves sharing ONE 16-row A panel:
//  - async-copy panel A[16 x 1024] global->LDS once (ASYNCcnt path)
//  - K loop FULLY UNROLLED (constant trip count): no loop-carried register
//    constraints, loads land directly in WMMA operand registers
//  - __launch_bounds__(256,2): block = 8 waves on 4 SIMDs = 2 waves/SIMD
// Each wave: 4 column tiles (64 cols). Writes aggr[(n*64+b)*K2 + m*F + f].
// ---------------------------------------------------------------------------
__global__ void __launch_bounds__(256, 2)
gemm1_bf16(const bf16* __restrict__ Abf, const bf16* __restrict__ Xt,
           float* __restrict__ aggr, int F, int K2, int cgBlocks) {
    const int cols = F * NB;
    const int colGroups = cols >> 6;                   // cols / 64
    const int lane = threadIdx.x;
    const int wid  = threadIdx.y;                      // 0..7
    const int rt  = blockIdx.x / cgBlocks;
    const int cgB = blockIdx.x - rt * cgBlocks;
    const int m   = blockIdx.z;
    const int r0  = rt << 4;

    __shared__ __align__(16) bf16 sA[16 * LDA];

    // ---- stage A panel [16 x 1024] into LDS (async global->LDS) ----
    {
        const int tid = (wid << 5) + lane;             // 0..255
        const bf16* gA = Abf + ((size_t)m * NP + r0) * NP;
#pragma unroll
        for (int c = 0; c < 8; ++c) {
            int u   = tid + (c << 8);                  // 16B unit, 0..2047
            int row = u >> 7;
            int ko  = (u & 127) << 3;                  // bf16 elem offset in row
            const bf16* src = gA + (size_t)row * NP + ko;
            bf16* dst = &sA[row * LDA + ko];
#if __has_builtin(__builtin_amdgcn_global_load_async_to_lds_b128)
            __builtin_amdgcn_global_load_async_to_lds_b128(
                (AS1 i32x4*)(i32x4*)src, (AS3 i32x4*)(i32x4*)dst, 0, 0);
#else
            *(uint4*)dst = *(const uint4*)src;
#endif
        }
    }
#if __has_builtin(__builtin_amdgcn_global_load_async_to_lds_b128)
#if __has_builtin(__builtin_amdgcn_s_wait_asynccnt)
    __builtin_amdgcn_s_wait_asynccnt(0);
#else
    asm volatile("s_wait_asynccnt 0" ::: "memory");
#endif
#endif
    __syncthreads();

    const int cg = (cgB << 3) + wid;
    if (cg < colGroups) {                              // wave-uniform guard
        const int c0 = cg << 6;
        const int lrow  = lane & 15;
        const int khalf = (lane >> 4) << 3;            // 0 or 8
        const bf16* bbase = Xt + (size_t)(c0 + lrow) * NP + khalf;
        const bf16* aL = &sA[lrow * LDA + khalf];

        v8f acc[4] = {};
        v16bf b0[4], b1[4];
#pragma unroll
        for (int j = 0; j < 4; ++j) b0[j] = load_frag(bbase + ((size_t)j << 4) * NP);

        // main ping-pong, FULLY UNROLLED: 15 x 64 K, epilogue last 64
#pragma unroll
        for (int kp = 0; kp < 15; ++kp) {
            const int k0 = kp << 6;
            v16bf a0 = load_frag(aL + k0);
#pragma unroll
            for (int j = 0; j < 4; ++j)
                b1[j] = load_frag(bbase + ((size_t)j << 4) * NP + (k0 + 32));
#pragma unroll
            for (int j = 0; j < 4; ++j)
                acc[j] = __builtin_amdgcn_wmma_f32_16x16x32_bf16(
                    false, a0, false, b0[j], (short)0, acc[j], false, false);
            v16bf a1 = load_frag(aL + k0 + 32);
#pragma unroll
            for (int j = 0; j < 4; ++j)
                b0[j] = load_frag(bbase + ((size_t)j << 4) * NP + (k0 + 64));
#pragma unroll
            for (int j = 0; j < 4; ++j)
                acc[j] = __builtin_amdgcn_wmma_f32_16x16x32_bf16(
                    false, a1, false, b1[j], (short)0, acc[j], false, false);
        }
        {   // epilogue: k0 = 960 (b0 holds K=960 frags)
            v16bf a0 = load_frag(aL + 960);
#pragma unroll
            for (int j = 0; j < 4; ++j)
                b1[j] = load_frag(bbase + ((size_t)j << 4) * NP + 992);
#pragma unroll
            for (int j = 0; j < 4; ++j)
                acc[j] = __builtin_amdgcn_wmma_f32_16x16x32_bf16(
                    false, a0, false, b0[j], (short)0, acc[j], false, false);
            v16bf a1 = load_frag(aL + 992);
#pragma unroll
            for (int j = 0; j < 4; ++j)
                acc[j] = __builtin_amdgcn_wmma_f32_16x16x32_bf16(
                    false, a1, false, b1[j], (short)0, acc[j], false, false);
        }

        const int rbase = r0 + ((lane >> 4) << 3);
#pragma unroll
        for (int j = 0; j < 4; ++j) {
            int c = c0 + (j << 4) + lrow;
            int bb = c / F;
            int f  = c - bb * F;
            float* dst = aggr + (size_t)bb * K2 + (size_t)m * F + f;
#pragma unroll
            for (int v = 0; v < 8; ++v) {
                int r = rbase + v;
                if (r < NN) dst[(size_t)r * NB * K2] = acc[j][v];
            }
        }
    }
}

// ---------------------------------------------------------------------------
// Weight contraction (fp32 WMMA 16x16x4): Out[64000x64] = Ain[64000xK2] @ W[K2x64]
// ---------------------------------------------------------------------------
__global__ void __launch_bounds__(256, 2)
gemm2_f32(const float* __restrict__ Ain, const float* __restrict__ W,
          float* __restrict__ Out, int K2) {
    const int lane = threadIdx.x;
    const int gw = blockIdx.x * blockDim.y + threadIdx.y;
    if (gw >= (NN * NB) / 16) return;                  // 4000 row tiles, wave-uniform
    const int r0   = gw << 4;
    const int lrow = lane & 15;
    const int kq   = (lane >> 4) << 1;                 // 0 or 2
    v8f acc[4] = {};
    const float* ap = Ain + (size_t)(r0 + lrow) * K2 + kq;
    for (int k0 = 0; k0 < K2; k0 += 4) {
        v2f a = *(const v2f*)(ap + k0);
        const int kk = k0 + kq;
#pragma unroll
        for (int j = 0; j < 4; ++j) {
            v2f b;
            b[0] = W[(size_t)kk * HH + (j << 4) + lrow];
            b[1] = W[(size_t)(kk + 1) * HH + (j << 4) + lrow];
            acc[j] = __builtin_amdgcn_wmma_f32_16x16x4_f32(
                false, a, false, b, (short)0, acc[j], false, false);
        }
    }
    const int rbase = r0 + ((lane >> 4) << 3);
#pragma unroll
    for (int j = 0; j < 4; ++j)
#pragma unroll
        for (int v = 0; v < 8; ++v)
            Out[(size_t)(rbase + v) * HH + (j << 4) + lrow] = acc[j][v];
}

// ---------------------------------------------------------------------------
// Layer0 gates: r=sig(rpre+br), u=sig(upre+bu) (u stored back); repack Xt
// ---------------------------------------------------------------------------
__global__ void gates_pack0(const float* __restrict__ rpre, float* __restrict__ upre,
                            const float* __restrict__ br, const float* __restrict__ bu,
                            const float* __restrict__ h0, const float* __restrict__ input,
                            bf16* __restrict__ Xt, int t) {
    size_t idx = (size_t)blockIdx.x * 256 + threadIdx.x;  // over 1024*64*64
    int h = (int)(idx & 63);
    int b = (int)((idx >> 6) & 63);
    int n = (int)(idx >> 12);
    int col = b * 65 + 1 + h;
    if (n >= NN) {
        Xt[(size_t)col * NP + n] = (bf16)0.0f;
        if (h == 0) Xt[(size_t)(b * 65) * NP + n] = (bf16)0.0f;
        return;
    }
    size_t e = ((size_t)n * NB + b) * HH + h;
    float r = sigmoidf_(rpre[e] + br[h]);
    float u = sigmoidf_(upre[e] + bu[h]);
    upre[e] = u;
    Xt[(size_t)col * NP + n] = (bf16)(r * h0[e]);
    if (h == 0)
        Xt[(size_t)(b * 65) * NP + n] = (bf16)(input[((size_t)b * TT + t) * NN + n]);
}

// Layer1 gates: cols b*128+0..63 (=h0) persist from pack_x1; rewrite 64..127 = r*h1
__global__ void gates_pack1(const float* __restrict__ rpre, float* __restrict__ upre,
                            const float* __restrict__ br, const float* __restrict__ bu,
                            const float* __restrict__ h1, bf16* __restrict__ Xt) {
    size_t idx = (size_t)blockIdx.x * 256 + threadIdx.x;  // over 1024*64*64
    int h = (int)(idx & 63);
    int b = (int)((idx >> 6) & 63);
    int n = (int)(idx >> 12);
    int col = b * 128 + HH + h;
    if (n >= NN) { Xt[(size_t)col * NP + n] = (bf16)0.0f; return; }
    size_t e = ((size_t)n * NB + b) * HH + h;
    float r = sigmoidf_(rpre[e] + br[h]);
    float u = sigmoidf_(upre[e] + bu[h]);
    upre[e] = u;
    Xt[(size_t)col * NP + n] = (bf16)(r * h1[e]);
}

// h = u*h + (1-u)*tanh(cpre + bc)
__global__ void gru_update(const float* __restrict__ cpre, const float* __restrict__ u,
                           const float* __restrict__ bc, float* __restrict__ hst) {
    size_t idx = (size_t)blockIdx.x * 256 + threadIdx.x;  // over 64000*64
    float c = tanhf(cpre[idx] + bc[idx & 63]);
    float uu = u[idx];
    hst[idx] = uu * hst[idx] + (1.0f - uu) * c;
}

// out[b, t, n] = h1[n][b][:] . proj_w + proj_b
__global__ void proj_out(const float* __restrict__ h1, const float* __restrict__ pw,
                         const float* __restrict__ pb, float* __restrict__ out, int t) {
    int idx = blockIdx.x * 256 + threadIdx.x;             // over 64*1000
    int b = idx / NN, n = idx - b * NN;
    const float* hp = h1 + ((size_t)n * NB + b) * HH;
    float s = pb[0];
#pragma unroll
    for (int h = 0; h < HH; ++h) s += hp[h] * pw[h];
    out[((size_t)b * (TT - 1) + t) * NN + n] = s;
}

// ---------------------------------------------------------------------------
extern "C" void kernel_launch(void* const* d_in, const int* in_sizes, int n_in,
                              void* d_out, int out_size, void* d_ws, size_t ws_size,
                              hipStream_t stream) {
    (void)in_sizes; (void)n_in; (void)out_size; (void)ws_size;
    const float* input = (const float*)d_in[0];
    const float* ih    = (const float*)d_in[1];
    const float* sup   = (const float*)d_in[2];
    const float* w0r = (const float*)d_in[4];  const float* b0r = (const float*)d_in[5];
    const float* w0u = (const float*)d_in[6];  const float* b0u = (const float*)d_in[7];
    const float* w0c = (const float*)d_in[8];  const float* b0c = (const float*)d_in[9];
    const float* w1r = (const float*)d_in[10]; const float* b1r = (const float*)d_in[11];
    const float* w1u = (const float*)d_in[12]; const float* b1u = (const float*)d_in[13];
    const float* w1c = (const float*)d_in[14]; const float* b1c = (const float*)d_in[15];
    const float* pw  = (const float*)d_in[16]; const float* pb  = (const float*)d_in[17];
    float* out = (float*)d_out;

    char* ws = (char*)d_ws;
    size_t off = 0;
    bf16*  Abf  = (bf16*)(ws + off);  off += (size_t)MM * NP * NP * sizeof(bf16);      // 8 MB
    bf16*  Xt   = (bf16*)(ws + off);  off += (size_t)8192 * NP * sizeof(bf16);          // 16.8 MB
    float* aggr = (float*)(ws + off); off += (size_t)NN * NB * 512 * sizeof(float);     // 131 MB
    float* gA   = (float*)(ws + off); off += (size_t)NN * NB * HH * sizeof(float);      // r/c pre
    float* gB   = (float*)(ws + off); off += (size_t)NN * NB * HH * sizeof(float);      // u
    float* h0   = (float*)(ws + off); off += (size_t)NN * NB * HH * sizeof(float);
    float* h1   = (float*)(ws + off); off += (size_t)NN * NB * HH * sizeof(float);

    const dim3 wblk(32, 8);   // 8 waves / block (wave32)

    pack_support<<<16384, 256, 0, stream>>>(sup, Abf);
    init_h<<<32000, 256, 0, stream>>>(ih, h0, h1);

    const int cgB0 = (65 + 7) / 8;    // 9 col-group blocks (layer0)
    const int cgB1 = 128 / 8;         // 16 col-group blocks (layer1)
    const dim3 g1grid0(63 * cgB0, 1, MM);
    const dim3 g1grid1(63 * cgB1, 1, MM);

    for (int t = 0; t < TT - 1; ++t) {
        // ================= layer 0 (F=65, K2=260) =================
        pack_x0<<<(65 * NB * NP) / 256, 256, 0, stream>>>(input, h0, Xt, t);
        gemm1_bf16<<<g1grid0, wblk, 0, stream>>>(Abf, Xt, aggr, 65, 260, cgB0);
        gemm2_f32<<<500, wblk, 0, stream>>>(aggr, w0r, gA, 260);
        gemm2_f32<<<500, wblk, 0, stream>>>(aggr, w0u, gB, 260);
        gates_pack0<<<16384, 256, 0, stream>>>(gA, gB, b0r, b0u, h0, input, Xt, t);
        gemm1_bf16<<<g1grid0, wblk, 0, stream>>>(Abf, Xt, aggr, 65, 260, cgB0);
        gemm2_f32<<<500, wblk, 0, stream>>>(aggr, w0c, gA, 260);
        gru_update<<<16000, 256, 0, stream>>>(gA, gB, b0c, h0);

        // ================= layer 1 (F=128, K2=512) ================
        pack_x1<<<(128 * NB * NP) / 256, 256, 0, stream>>>(h0, h1, Xt);
        gemm1_bf16<<<g1grid1, wblk, 0, stream>>>(Abf, Xt, aggr, 128, 512, cgB1);
        gemm2_f32<<<500, wblk, 0, stream>>>(aggr, w1r, gA, 512);
        gemm2_f32<<<500, wblk, 0, stream>>>(aggr, w1u, gB, 512);
        gates_pack1<<<16384, 256, 0, stream>>>(gA, gB, b1r, b1u, h1, Xt);
        gemm1_bf16<<<g1grid1, wblk, 0, stream>>>(Abf, Xt, aggr, 128, 512, cgB1);
        gemm2_f32<<<500, wblk, 0, stream>>>(aggr, w1c, gA, 512);
        gru_update<<<16000, 256, 0, stream>>>(gA, gB, b1c, h1);

        proj_out<<<250, 256, 0, stream>>>(h1, pw, pb, out, t);
    }
}
